// KernelizedAttention_14869176779022
// MI455X (gfx1250) — matrix operand order repaired and checked by reading, not compiled, not run on hardware
//
#include <hip/hip_runtime.h>

// out[8192,1024] = x[8192,1024] @ Wv[1024,1024]^T + bv    (the phi-attention s cancels: out == v)

typedef __attribute__((ext_vector_type(16))) __bf16 bf16x16;
typedef __attribute__((ext_vector_type(8)))  float  f32x8;

#define GM 8192   // B*S
#define GN 1024   // E
#define GK 1024   // E
#define TM 128
#define TN 128
#define TK 32
#define LDS_H 40  // halves per LDS row: 32 data + 8 pad = 80 bytes (16B aligned, bank-conflict-free)

union FragU {
    bf16x16 v;
    uint4   q[2];
};

__device__ __forceinline__ void cvt_split(float a, unsigned short& hi, unsigned short& lo) {
    // round-to-nearest-even f32 -> bf16 hi, residual -> bf16 lo
    unsigned int u = __float_as_uint(a);
    unsigned int r = u + 0x7FFFu + ((u >> 16) & 1u);
    hi = (unsigned short)(r >> 16);
    float hf = __uint_as_float(r & 0xFFFF0000u);
    float l  = a - hf;
    unsigned int ul = __float_as_uint(l);
    unsigned int rl = ul + 0x7FFFu + ((ul >> 16) & 1u);
    lo = (unsigned short)(rl >> 16);
}

__device__ __forceinline__ uint4 pack4(const unsigned short* s) {
    uint4 r;
    r.x = (unsigned int)s[0] | ((unsigned int)s[1] << 16);
    r.y = (unsigned int)s[2] | ((unsigned int)s[3] << 16);
    r.z = (unsigned int)s[4] | ((unsigned int)s[5] << 16);
    r.w = (unsigned int)s[6] | ((unsigned int)s[7] << 16);
    return r;
}

__global__ __launch_bounds__(256) void vproj_wmma_kernel(
    const float* __restrict__ x, const float* __restrict__ Wv,
    const float* __restrict__ bv, float* __restrict__ out)
{
    __shared__ __align__(16) unsigned short lAhi[TM * LDS_H];
    __shared__ __align__(16) unsigned short lAlo[TM * LDS_H];
    __shared__ __align__(16) unsigned short lBhi[TN * LDS_H];
    __shared__ __align__(16) unsigned short lBlo[TN * LDS_H];

    const int tid    = threadIdx.x;
    const int lane   = tid & 31;        // wave32
    const int wave   = tid >> 5;        // 0..7
    const int g      = lane >> 4;       // K-group within wave (WMMA layout)
    const int lr     = lane & 15;       // row/col within 16x16 tile
    const int waveM  = (wave >> 2) * 64;
    const int waveN  = (wave & 3) * 32;
    const int blockM = blockIdx.y * TM;
    const int blockN = blockIdx.x * TN;

    // staging: each thread loads 16 contiguous f32 of one row
    const int ldr = tid >> 1;           // 0..127
    const int ldh = (tid & 1) * 16;     // half-row offset (0 or 16)

    f32x8 acc[4][2];
    const f32x8 zero = {};
#pragma unroll
    for (int ms = 0; ms < 4; ++ms)
#pragma unroll
        for (int ns = 0; ns < 2; ++ns) acc[ms][ns] = zero;

    for (int k0 = 0; k0 < GK; k0 += TK) {
        // ---- stage A tile (x) and B tile (Wv, naturally [n][k]) as bf16 hi/lo ----
        {
            const float4* s4 = (const float4*)(x + (size_t)(blockM + ldr) * GK + k0 + ldh);
            float4 f0 = s4[0], f1 = s4[1], f2 = s4[2], f3 = s4[3];
            float vals[16] = {f0.x, f0.y, f0.z, f0.w, f1.x, f1.y, f1.z, f1.w,
                              f2.x, f2.y, f2.z, f2.w, f3.x, f3.y, f3.z, f3.w};
            unsigned short hi[16], lo[16];
#pragma unroll
            for (int j = 0; j < 16; ++j) cvt_split(vals[j], hi[j], lo[j]);
            uint4* dH = (uint4*)&lAhi[ldr * LDS_H + ldh];
            uint4* dL = (uint4*)&lAlo[ldr * LDS_H + ldh];
            dH[0] = pack4(hi);     dH[1] = pack4(hi + 8);
            dL[0] = pack4(lo);     dL[1] = pack4(lo + 8);
        }
        {
            const float4* s4 = (const float4*)(Wv + (size_t)(blockN + ldr) * GK + k0 + ldh);
            float4 f0 = s4[0], f1 = s4[1], f2 = s4[2], f3 = s4[3];
            float vals[16] = {f0.x, f0.y, f0.z, f0.w, f1.x, f1.y, f1.z, f1.w,
                              f2.x, f2.y, f2.z, f2.w, f3.x, f3.y, f3.z, f3.w};
            unsigned short hi[16], lo[16];
#pragma unroll
            for (int j = 0; j < 16; ++j) cvt_split(vals[j], hi[j], lo[j]);
            uint4* dH = (uint4*)&lBhi[ldr * LDS_H + ldh];
            uint4* dL = (uint4*)&lBlo[ldr * LDS_H + ldh];
            dH[0] = pack4(hi);     dH[1] = pack4(hi + 8);
            dL[0] = pack4(lo);     dL[1] = pack4(lo + 8);
        }
        if (k0 + TK < GK) {
            __builtin_prefetch(x  + (size_t)(blockM + ldr) * GK + k0 + TK + ldh, 0, 0);
            __builtin_prefetch(Wv + (size_t)(blockN + ldr) * GK + k0 + TK + ldh, 0, 0);
        }
        __syncthreads();

        // ---- load WMMA fragments (ISA 16-bit wave32 layouts), 2x b128 each ----
        FragU aHi[4], aLo[4], bHi[2], bLo[2];
#pragma unroll
        for (int ms = 0; ms < 4; ++ms) {
            const uint4* p  = (const uint4*)&lAhi[(waveM + ms * 16 + lr) * LDS_H];
            const uint4* pl = (const uint4*)&lAlo[(waveM + ms * 16 + lr) * LDS_H];
            // A row striped in-lane: g=0 -> K0-7,K16-23 ; g=1 -> K8-15,K24-31
            aHi[ms].q[0] = p[g];      aHi[ms].q[1] = p[2 + g];
            aLo[ms].q[0] = pl[g];     aLo[ms].q[1] = pl[2 + g];
        }
#pragma unroll
        for (int ns = 0; ns < 2; ++ns) {
            const uint4* p  = (const uint4*)&lBhi[(waveN + ns * 16 + lr) * LDS_H];
            const uint4* pl = (const uint4*)&lBlo[(waveN + ns * 16 + lr) * LDS_H];
            // B col striped in-lane: g=0 -> K0-7,K16-23 ; g=1 -> K8-15,K24-31
            bHi[ns].q[0] = p[g];      bHi[ns].q[1] = p[2 + g];
            bLo[ns].q[0] = pl[g];     bLo[ns].q[1] = pl[2 + g];
        }

        // ---- split-bf16 FMA: Ahi*Bhi + Ahi*Blo + Alo*Bhi, f32 accumulate ----
#pragma unroll
        for (int ms = 0; ms < 4; ++ms)
#pragma unroll
            for (int ns = 0; ns < 2; ++ns) {
                f32x8 c = acc[ms][ns];
                c = __builtin_amdgcn_wmma_f32_16x16x32_bf16(
                        false, aHi[ms].v, false, bHi[ns].v, (short)0, c, false, false);
                c = __builtin_amdgcn_wmma_f32_16x16x32_bf16(
                        false, aHi[ms].v, false, bLo[ns].v, (short)0, c, false, false);
                c = __builtin_amdgcn_wmma_f32_16x16x32_bf16(
                        false, aLo[ms].v, false, bHi[ns].v, (short)0, c, false, false);
                acc[ms][ns] = c;
            }
        __syncthreads();
    }

    // ---- epilogue: add bias, store f32. D layout: VGPR r -> row r + 8*g, col = lane&15 ----
#pragma unroll
    for (int ms = 0; ms < 4; ++ms)
#pragma unroll
        for (int ns = 0; ns < 2; ++ns) {
            const int col  = blockN + waveN + ns * 16 + lr;
            const float bb = bv[col];
            float* po = out + (size_t)(blockM + waveM + ms * 16 + 8 * g) * GN + col;
#pragma unroll
            for (int r = 0; r < 8; ++r)
                po[(size_t)r * GN] = acc[ms][ns][r] + bb;
        }
}

extern "C" void kernel_launch(void* const* d_in, const int* in_sizes, int n_in,
                              void* d_out, int out_size, void* d_ws, size_t ws_size,
                              hipStream_t stream) {
    (void)in_sizes; (void)n_in; (void)out_size; (void)d_ws; (void)ws_size;
    // setup_inputs order: x, Wq, bq, Wk, bk, Wv, bv, weights
    const float* x  = (const float*)d_in[0];
    const float* Wv = (const float*)d_in[5];
    const float* bv = (const float*)d_in[6];
    float* out = (float*)d_out;

    dim3 grid(GN / TN, GM / TM);   // (8, 64) = 512 workgroups
    vproj_wmma_kernel<<<grid, 256, 0, stream>>>(x, Wv, bv, out);
}
